// MultiHeadAttention_30751965839713
// MI455X (gfx1250) — compile-verified, matrix operand order checked by reading, not run on hardware
//
#include <hip/hip_runtime.h>

#define N_TOK 2048
#define DIM   128
#define NHEAD 40
#define HPG   5     // heads per group (attn)
#define NHG   8     // head groups

typedef __attribute__((ext_vector_type(8)))  __bf16 v8bf;
typedef __attribute__((ext_vector_type(16))) __bf16 v16bf;
typedef __attribute__((ext_vector_type(8)))  float  v8f;
typedef __attribute__((ext_vector_type(4)))  unsigned int v4u;
typedef __attribute__((ext_vector_type(8)))  int v8i;
typedef __attribute__((ext_vector_type(4)))  int v4i;

static __device__ __forceinline__ v8f zero8() {
  v8f z;
#pragma unroll
  for (int i = 0; i < 8; ++i) z[i] = 0.0f;
  return z;
}

static __device__ __forceinline__ v16bf cat8(v8bf a, v8bf b) {
  v16bf r;
#pragma unroll
  for (int i = 0; i < 8; ++i) { r[i] = a[i]; r[i + 8] = b[i]; }
  return r;
}

// A/B fragment (16-bit, 16x32 / 32x16) from contiguous bf16 row:
// elems 0..7  = K { g*8 .. g*8+7 }, elems 8..15 = K { 16+g*8 .. }
static __device__ __forceinline__ v16bf frag_bf16(const __bf16* p, int g) {
  v8bf a = *(const v8bf*)(p + g * 8);
  v8bf b = *(const v8bf*)(p + 16 + g * 8);
  return cat8(a, b);
}

// same fragment from contiguous fp32 (converted to bf16)
static __device__ __forceinline__ v16bf frag_f32(const float* p, int g) {
  const float4* pa = (const float4*)(p + g * 8);
  const float4* pb = (const float4*)(p + 16 + g * 8);
  float4 x0 = pa[0], x1 = pa[1], y0 = pb[0], y1 = pb[1];
  v16bf r;
  r[0] = (__bf16)x0.x; r[1] = (__bf16)x0.y; r[2]  = (__bf16)x0.z; r[3]  = (__bf16)x0.w;
  r[4] = (__bf16)x1.x; r[5] = (__bf16)x1.y; r[6]  = (__bf16)x1.z; r[7]  = (__bf16)x1.w;
  r[8] = (__bf16)y0.x; r[9] = (__bf16)y0.y; r[10] = (__bf16)y0.z; r[11] = (__bf16)y0.w;
  r[12] = (__bf16)y1.x; r[13] = (__bf16)y1.y; r[14] = (__bf16)y1.z; r[15] = (__bf16)y1.w;
  return r;
}

static __device__ __forceinline__ v8f wmma_bf16(v16bf a, v16bf b, v8f c) {
  return __builtin_amdgcn_wmma_f32_16x16x32_bf16(false, a, false, b, (short)0, c,
                                                 false, false);
}

// ------------------------------------------------------------------ TDM
// 2D tile load via Tensor Data Mover (D# per CDNA5 ISA ch.8).
// Units are elements of 2 bytes (data_size code 1). Tracked by TENSORcnt.
static __device__ __forceinline__ void tdm_load_2d(const void* gptr,
                                                   unsigned lds_byte,
                                                   unsigned tile_d0,
                                                   unsigned tile_d1,
                                                   unsigned tensor_d0,
                                                   unsigned tensor_d1,
                                                   unsigned stride_d0) {
  unsigned long long ga = (unsigned long long)(size_t)gptr;
  v4u g0;
  g0[0] = 1u;                                          // count=1, user mode
  g0[1] = lds_byte;                                    // lds_addr
  g0[2] = (unsigned)(ga & 0xffffffffu);                // global_addr[31:0]
  g0[3] = (unsigned)((ga >> 32) & 0x01ffffffu)         // global_addr[56:32]
          | (2u << 30);                                // type = 2 ("image")
  v8i g1;
  g1[0] = (int)(1u << 16);                             // data_size = 2 bytes
  g1[1] = (int)((tensor_d0 & 0xffffu) << 16);          // tensor_dim0[15:0]
  g1[2] = (int)(((tensor_d0 >> 16) & 0xffffu) |        // tensor_dim0[31:16]
                ((tensor_d1 & 0xffffu) << 16));        // tensor_dim1[15:0]
  g1[3] = (int)(((tensor_d1 >> 16) & 0xffffu) |        // tensor_dim1[31:16]
                (tile_d0 << 16));                      // tile_dim0
  g1[4] = (int)(tile_d1 & 0xffffu);                    // tile_dim1, tile_dim2=0
  g1[5] = (int)stride_d0;                              // tensor_dim0_stride lo
  g1[6] = 0;
  g1[7] = 0;
  v4i gz4 = {0, 0, 0, 0};
  v8i gz8 = {0, 0, 0, 0, 0, 0, 0, 0};
  __builtin_amdgcn_tensor_load_to_lds(g0, g1, gz4, gz4, gz8, 0);
}

// ---------------------------------------------------------------- kernel 0
__global__ __launch_bounds__(256) void cvt_x_kernel(const float* __restrict__ X,
                                                    __bf16* __restrict__ Xb) {
  int i = blockIdx.x * 256 + threadIdx.x;   // grid covers exactly N*D
  Xb[i] = (__bf16)X[i];
}

// ---------------------------------------------------------------- kernel 1
// grid (N/16, H, 2); block = 1 wave.
// z==0: Q[h] = X @ W[h]       -> Qb  [H][N][D] bf16 row-major
// z==1: XVt[h] = (X @ V[h])^T -> [H][D][N] bf16  (+ colsum[h][e])
__global__ __launch_bounds__(32) void proj_kernel(
    const float* __restrict__ X, const float* __restrict__ W,
    const float* __restrict__ V, __bf16* __restrict__ Qb,
    __bf16* __restrict__ XVt, float* __restrict__ colsum) {
  const int rblk = blockIdx.x;
  const int h    = blockIdx.y;
  const int isV  = blockIdx.z;
  const int lane = threadIdx.x;
  const int g = lane >> 4, n = lane & 15;
  const int r0 = rblk * 16;

  const float* M = (isV ? V : W) + (size_t)h * DIM * DIM;

  __shared__ __align__(16) float Wt[DIM * 16];   // [k][e-tile] fp32, 8KB

  v16bf xa[4];
  const float* xrow = X + (size_t)(r0 + n) * DIM;
#pragma unroll
  for (int kc = 0; kc < 4; ++kc) xa[kc] = frag_f32(xrow + kc * 32, g);

  for (int e0 = 0; e0 < DIM; e0 += 16) {
    __syncthreads();
#pragma unroll
    for (int rr = 0; rr < 4; ++rr) {
      int k = lane + rr * 32;
      const float4* src = (const float4*)(M + (size_t)k * DIM + e0);
      float4* dst = (float4*)(Wt + k * 16);
#pragma unroll
      for (int q = 0; q < 4; ++q) dst[q] = src[q];
    }
    __syncthreads();

    v8f acc = zero8();
#pragma unroll
    for (int kc = 0; kc < 4; ++kc) {
      v16bf b;
#pragma unroll
      for (int j = 0; j < 8; ++j) {
        b[j]     = (__bf16)Wt[(kc * 32 + g * 8 + j) * 16 + n];
        b[j + 8] = (__bf16)Wt[(kc * 32 + 16 + g * 8 + j) * 16 + n];
      }
      acc = wmma_bf16(xa[kc], b, acc);
    }

    if (!isV) {
      __bf16* out = Qb + ((size_t)h * N_TOK + r0) * DIM + e0 + n;
#pragma unroll
      for (int i = 0; i < 8; ++i) out[(size_t)(g * 8 + i) * DIM] = (__bf16)acc[i];
    } else {
      v8bf pk;
      float s = 0.0f;
#pragma unroll
      for (int i = 0; i < 8; ++i) { pk[i] = (__bf16)acc[i]; s += acc[i]; }
      *(v8bf*)(XVt + ((size_t)h * DIM + e0 + n) * N_TOK + r0 + g * 8) = pk;
      atomicAdd(&colsum[h * DIM + e0 + n], s);
    }
  }
}

// ---------------------------------------------------------------- kernel 2
// grid (N/64, NHG); block = 128 (4 waves). Wave w owns rows r0..r0+15 and
// loops over HPG heads with flash-attention online softmax. K/V tiles are
// double-buffered in LDS and filled by the Tensor Data Mover (each wave
// issues its own slice; uniform control flow, per-wave TENSORcnt waits).
__global__ __launch_bounds__(128) void attn_kernel(
    const __bf16* __restrict__ Xb, const __bf16* __restrict__ Qb,
    const __bf16* __restrict__ XVt, const float* __restrict__ colsum,
    float* __restrict__ out) {
  const int tid  = threadIdx.x;
  const int wave = tid >> 5;
  const int lane = tid & 31;
  const int g = lane >> 4, n = lane & 15;
  const int r0 = blockIdx.x * 64 + wave * 16;

  __shared__ __align__(16) __bf16 Kt[2][32 * DIM];   // [buf][key][d]  2x8KB
  __shared__ __align__(16) __bf16 Vt[2][DIM * 32];   // [buf][d][key]  2x8KB
  __shared__ __align__(16) __bf16 Pb[4][512];        // per-wave P 16x32

  __bf16* pw = Pb[wave];
  const float scale = 0.0883883476483184f;   // 1/sqrt(128)

  float Oacc[8][8];
#pragma unroll
  for (int f = 0; f < 8; ++f)
#pragma unroll
    for (int i = 0; i < 8; ++i) Oacc[f][i] = 0.0f;

  for (int hi = 0; hi < HPG; ++hi) {
    const int h = blockIdx.y * HPG + hi;
    const __bf16* xvh = XVt + (size_t)h * DIM * N_TOK;

    v16bf qa[4];
    const __bf16* qrow = Qb + ((size_t)h * N_TOK + r0 + n) * DIM;
#pragma unroll
    for (int kc = 0; kc < 4; ++kc) qa[kc] = frag_bf16(qrow + kc * 32, g);

    v8f O[8];
#pragma unroll
    for (int f = 0; f < 8; ++f) O[f] = zero8();
    float mrow[8], lrow[8];
#pragma unroll
    for (int i = 0; i < 8; ++i) { mrow[i] = -1e30f; lrow[i] = 0.0f; }

    // TDM issue: wave w loads K rows [j0+8w, j0+8w+8) and V rows [32w, 32w+32)
    auto issue_tile = [&](int j0, int b) {
      tdm_load_2d(Xb + (size_t)(j0 + wave * 8) * DIM,
                  (unsigned)(size_t)(void*)&Kt[b][wave * 1024],
                  /*tile*/ 1024, 1, /*tensor*/ (unsigned)(N_TOK * DIM), 1,
                  /*stride*/ (unsigned)(N_TOK * DIM));
      tdm_load_2d(xvh + (size_t)(wave * 32) * N_TOK + j0,
                  (unsigned)(size_t)(void*)&Vt[b][wave * 1024],
                  /*tile*/ 32, 32, /*tensor*/ (unsigned)N_TOK, DIM,
                  /*stride*/ (unsigned)N_TOK);
    };

    issue_tile(0, 0);
    for (int j0 = 0; j0 < N_TOK; j0 += 32) {
      const int buf = (j0 >> 5) & 1;
      if (j0 + 32 < N_TOK) {
        issue_tile(j0 + 32, buf ^ 1);            // prefetch next tile pair
        __builtin_amdgcn_s_wait_tensorcnt(2);    // current pair complete
      } else {
        __builtin_amdgcn_s_wait_tensorcnt(0);
      }
      __syncthreads();                           // all slices visible

      const __bf16* K = Kt[buf];
      const __bf16* Vv = Vt[buf];

      // ---- S = Q K^T  (two 16x16 tiles over 32 keys)
      v8f s0 = zero8(), s1 = zero8();
#pragma unroll
      for (int kc = 0; kc < 4; ++kc) {
        v16bf b0 = frag_bf16(K + (size_t)n * DIM + kc * 32, g);
        v16bf b1 = frag_bf16(K + (size_t)(16 + n) * DIM + kc * 32, g);
        s0 = wmma_bf16(qa[kc], b0, s0);
        s1 = wmma_bf16(qa[kc], b1, s1);
      }

      // ---- online softmax over the 32 new columns
      float p0[8], p1[8], alpha[8];
#pragma unroll
      for (int i = 0; i < 8; ++i) {
        float a = s0[i] * scale, b = s1[i] * scale;
        float mx = fmaxf(a, b);
#pragma unroll
        for (int msk = 8; msk >= 1; msk >>= 1)
          mx = fmaxf(mx, __shfl_xor(mx, msk, 32));
        float mnew = fmaxf(mrow[i], mx);
        float e0 = __expf(a - mnew);
        float e1 = __expf(b - mnew);
        p0[i] = e0; p1[i] = e1;
        float rs = e0 + e1;
#pragma unroll
        for (int msk = 8; msk >= 1; msk >>= 1)
          rs += __shfl_xor(rs, msk, 32);
        float al = __expf(mrow[i] - mnew);
        lrow[i] = lrow[i] * al + rs;
        mrow[i] = mnew;
        alpha[i] = al;
      }
#pragma unroll
      for (int f = 0; f < 8; ++f)
#pragma unroll
        for (int i = 0; i < 8; ++i) O[f][i] *= alpha[i];

      // ---- P (C layout) -> LDS -> A fragment
#pragma unroll
      for (int i = 0; i < 8; ++i) {
        int m = g * 8 + i;
        pw[m * 32 + n]      = (__bf16)p0[i];
        pw[m * 32 + 16 + n] = (__bf16)p1[i];
      }
      __syncthreads();
      v16bf pa = frag_bf16(pw + n * 32, g);

      // ---- O += P @ XV  (8 n-tiles of D)
#pragma unroll
      for (int f = 0; f < 8; ++f) {
        v16bf vb = frag_bf16(Vv + (f * 16 + n) * 32, g);
        O[f] = wmma_bf16(pa, vb, O[f]);
      }
      __syncthreads();                           // release buf for TDM reuse
    }

    // finalize this head: softmax normalize + (1e-8 * colsum) term
#pragma unroll
    for (int f = 0; f < 8; ++f) {
      float cs = colsum[h * DIM + f * 16 + n] * 1e-8f;
#pragma unroll
      for (int i = 0; i < 8; ++i) Oacc[f][i] += O[f][i] / lrow[i] + cs;
    }
  }

  // one atomic accumulation pass per block (8 head-groups sum into out)
#pragma unroll
  for (int f = 0; f < 8; ++f)
#pragma unroll
    for (int i = 0; i < 8; ++i)
      atomicAdd(&out[(size_t)(r0 + g * 8 + i) * DIM + f * 16 + n], Oacc[f][i]);
}

// ---------------------------------------------------------------- launch
extern "C" void kernel_launch(void* const* d_in, const int* in_sizes, int n_in,
                              void* d_out, int out_size, void* d_ws, size_t ws_size,
                              hipStream_t stream) {
  const float* X = (const float*)d_in[0];
  const float* W = (const float*)d_in[1];
  const float* V = (const float*)d_in[2];
  float* out = (float*)d_out;

  const size_t XB_BYTES = (size_t)N_TOK * DIM * 2;           // 512 KB
  const size_t QB_BYTES = (size_t)NHEAD * N_TOK * DIM * 2;   // 20 MB
  char* ws = (char*)d_ws;
  __bf16* Xb     = (__bf16*)ws;
  __bf16* Qb     = (__bf16*)(ws + XB_BYTES);
  __bf16* XVt    = (__bf16*)(ws + XB_BYTES + QB_BYTES);
  float*  colsum = (float*)(ws + XB_BYTES + 2 * QB_BYTES);

  (void)hipMemsetAsync(colsum, 0, (size_t)NHEAD * DIM * sizeof(float), stream);
  (void)hipMemsetAsync(out, 0, (size_t)N_TOK * DIM * sizeof(float), stream);

  cvt_x_kernel<<<dim3((N_TOK * DIM) / 256), dim3(256), 0, stream>>>(X, Xb);
  proj_kernel<<<dim3(N_TOK / 16, NHEAD, 2), dim3(32), 0, stream>>>(X, W, V, Qb,
                                                                   XVt, colsum);
  attn_kernel<<<dim3(N_TOK / 64, NHG), dim3(128), 0, stream>>>(Xb, Qb, XVt,
                                                               colsum, out);
}
